// CPPrefillForExport_66864050864552
// MI455X (gfx1250) — matrix operand order repaired and checked by reading, not compiled, not run on hardware
//
#include <hip/hip_runtime.h>
#include <hip/hip_bf16.h>

// ---------------- model constants ----------------
constexpr int L  = 5;
constexpr int D  = 1024;
constexpr int H  = 16;
constexpr int KV = 8;
constexpr int HD = 128;
constexpr int FF = 3072;
constexpr int T  = 2048;
constexpr int E  = H * HD;     // 2048
constexpr float EPS = 1e-6f;
constexpr float SCALE = 0.08838834764831845f;   // HD^-0.5
constexpr float NEG_BIG = -1e30f;

typedef unsigned short u16;
typedef __attribute__((ext_vector_type(16))) __bf16 v16bf;
typedef __attribute__((ext_vector_type(8)))  float  v8f;

struct __attribute__((may_alias)) alignas(16) Half8 { unsigned int u[4]; };
struct BFrag { Half8 lo; Half8 hi; };   // 32 bytes = 16 bf16

// CDNA5 async global->LDS (ASYNCcnt-tracked), guarded so a toolchain without
// the builtin falls back to an explicit VGPR-staged copy.
#if defined(__has_builtin)
#if __has_builtin(__builtin_amdgcn_global_load_async_to_lds_b128)
#define USE_ASYNC_LDS 1
#endif
#endif

#ifdef USE_ASYNC_LDS
typedef __attribute__((__vector_size__(16))) int v4i;
typedef __attribute__((address_space(1))) v4i gv4i;
typedef __attribute__((address_space(3))) v4i lv4i;
static __device__ __forceinline__ void async_copy16(const void* g, void* l) {
  __builtin_amdgcn_global_load_async_to_lds_b128(
      (gv4i*)(size_t)g, (lv4i*)(unsigned)(size_t)l, 0, 0);
}
static __device__ __forceinline__ void wait_async() {
#if __has_builtin(__builtin_amdgcn_s_wait_asynccnt)
  __builtin_amdgcn_s_wait_asynccnt(0);
#else
  asm volatile("s_wait_asynccnt 0x0" ::: "memory");
#endif
}
#endif

static __device__ __forceinline__ u16 f2bf(float x) {
  __bf16 b = (__bf16)x;            // hardware f32->bf16 (RNE) on gfx1250
  return __builtin_bit_cast(u16, b);
}

static __device__ __forceinline__ v16bf load_frag(const u16* p0, const u16* p1) {
  BFrag f;
  f.lo = *(const Half8*)p0;
  f.hi = *(const Half8*)p1;
  return __builtin_bit_cast(v16bf, f);
}

static __device__ __forceinline__ v8f wmma_bf16(v16bf a, v16bf b, v8f c) {
  return __builtin_amdgcn_wmma_f32_16x16x32_bf16(false, a, false, b, (short)0, c,
                                                 false, false);
}

// ---------------- GEMM: C[M,N] (+)= A[M,K](bf16) * W[N,K](f32->bf16)^T -------
// BM=BN=128, BK=32; 8 waves, each wave 64x32 (4x2 16x16 accumulators).
__global__ __launch_bounds__(256)
void gemm_bf16(const u16* __restrict__ A, const float* __restrict__ W,
               float* __restrict__ C, int M, int N, int K, int mode) {
  __shared__ alignas(16) u16 As[128 * 32];
  __shared__ alignas(16) u16 Bs[128 * 32];

  const int tid  = threadIdx.x;
  const int lane = tid & 31;
  const int wave = tid >> 5;
  const int hi   = lane >> 4;
  const int ln   = lane & 15;
  const int bm   = blockIdx.y * 128;
  const int bn   = blockIdx.x * 128;
  const int wm   = (wave >> 2) * 64;
  const int wn   = (wave & 3) * 32;

  const v8f vzero = {0.f, 0.f, 0.f, 0.f, 0.f, 0.f, 0.f, 0.f};
  v8f acc[4][2];
#pragma unroll
  for (int i = 0; i < 4; ++i)
#pragma unroll
    for (int j = 0; j < 2; ++j) acc[i][j] = vzero;

  const int frow = tid >> 1;
  const int fcol = (tid & 1) * 16;

  for (int kb = 0; kb < K; kb += 32) {
    __syncthreads();
    // fill A tile (bf16 -> bf16): async DMA straight into LDS when available
    {
      const u16* gsrc = A + (size_t)(bm + frow) * K + kb + fcol;
      u16* ldst = As + frow * 32 + fcol;
#ifdef USE_ASYNC_LDS
      async_copy16(gsrc, ldst);
      async_copy16(gsrc + 8, ldst + 8);
#else
      Half8 r0 = ((const Half8*)gsrc)[0];
      Half8 r1 = ((const Half8*)gsrc)[1];
      ((Half8*)ldst)[0] = r0;
      ((Half8*)ldst)[1] = r1;
#endif
      if (kb + 32 < K) __builtin_prefetch(gsrc + 32, 0, 1);
    }
    // fill B tile (f32 -> bf16)
    {
      const float* src = W + (size_t)(bn + frow) * K + kb + fcol;
      u16 tmp[16];
#pragma unroll
      for (int e = 0; e < 16; ++e) tmp[e] = f2bf(src[e]);
      Half8* dst = (Half8*)(Bs + frow * 32 + fcol);
      dst[0] = ((const Half8*)tmp)[0];
      dst[1] = ((const Half8*)tmp)[1];
      if (kb + 32 < K) __builtin_prefetch(src + 32, 0, 1);
    }
#ifdef USE_ASYNC_LDS
    wait_async();
#endif
    __syncthreads();

    v16bf afr[4], bfr[2];
#pragma unroll
    for (int i = 0; i < 4; ++i) {
      const u16* p = As + (wm + 16 * i + ln) * 32;
      afr[i] = load_frag(p + hi * 16, p + hi * 16 + 8);
    }
#pragma unroll
    for (int j = 0; j < 2; ++j) {
      const u16* p = Bs + (wn + 16 * j + ln) * 32 + hi * 16;
      bfr[j] = load_frag(p, p + 8);
    }
#pragma unroll
    for (int i = 0; i < 4; ++i)
#pragma unroll
      for (int j = 0; j < 2; ++j) acc[i][j] = wmma_bf16(afr[i], bfr[j], acc[i][j]);
  }

#pragma unroll
  for (int i = 0; i < 4; ++i)
#pragma unroll
    for (int j = 0; j < 2; ++j)
#pragma unroll
      for (int r = 0; r < 8; ++r) {
        int row = bm + wm + 16 * i + r + 8 * hi;
        int col = bn + wn + 16 * j + ln;
        size_t idx = (size_t)row * N + col;
        float v = acc[i][j][r];
        if (mode) C[idx] += v; else C[idx] = v;
      }
}

// ---------------- flash attention (bf16 wmma, f32 accum) ---------------------
// grid (T/128, H); each wave owns a 16-query tile, streams 32-key tiles.
__global__ __launch_bounds__(256)
void attn_kernel(const u16* __restrict__ qb, const u16* __restrict__ kb,
                 const u16* __restrict__ vtb, u16* __restrict__ outb) {
  __shared__ alignas(16) u16 Ps[8][16 * 32];

  const int head = blockIdx.y;
  const int kvh  = head >> 1;                 // G = 2
  const int wave = threadIdx.x >> 5;
  const int lane = threadIdx.x & 31;
  const int hi   = lane >> 4;
  const int ln   = lane & 15;
  const int q0   = blockIdx.x * 128 + wave * 16;

  // Q fragments: 16 rows x 128 K  -> 4 x (16x32) A-fragments
  v16bf qf[4];
  {
    const u16* qr = qb + ((size_t)(head * T + q0 + ln)) * HD;
#pragma unroll
    for (int kk = 0; kk < 4; ++kk)
      qf[kk] = load_frag(qr + kk * 32 + hi * 16, qr + kk * 32 + hi * 16 + 8);
  }

  const v8f vzero = {0.f, 0.f, 0.f, 0.f, 0.f, 0.f, 0.f, 0.f};
  v8f o[8];
#pragma unroll
  for (int nt = 0; nt < 8; ++nt) o[nt] = vzero;
  float mrun[8], lrun[8];
#pragma unroll
  for (int r = 0; r < 8; ++r) { mrun[r] = NEG_BIG; lrun[r] = 0.f; }

  const u16* kbase = kb + (size_t)kvh * T * HD;
  const u16* vbase = vtb + (size_t)kvh * HD * T;
  u16* pp = &Ps[wave][0];

  for (int jb = 0; jb < q0 + 16; jb += 32) {
    v8f s0 = vzero, s1 = vzero;
#pragma unroll
    for (int kk = 0; kk < 4; ++kk) {
      const u16* kr0 = kbase + (size_t)(jb + ln) * HD + kk * 32 + hi * 16;
      s0 = wmma_bf16(qf[kk], load_frag(kr0, kr0 + 8), s0);
      const u16* kr1 = kbase + (size_t)(jb + 16 + ln) * HD + kk * 32 + hi * 16;
      s1 = wmma_bf16(qf[kk], load_frag(kr1, kr1 + 8), s1);
    }

    float p0[8], p1[8];
#pragma unroll
    for (int r = 0; r < 8; ++r) {
      int row = q0 + r + 8 * hi;
      int c0 = jb + ln, c1 = c0 + 16;
      float a = (c0 <= row) ? s0[r] * SCALE : NEG_BIG;
      float b = (c1 <= row) ? s1[r] * SCALE : NEG_BIG;
      float mx = fmaxf(a, b);
#pragma unroll
      for (int off = 1; off < 16; off <<= 1) mx = fmaxf(mx, __shfl_xor(mx, off, 32));
      float mnew = fmaxf(mrun[r], mx);
      float alpha = __expf(mrun[r] - mnew);
      float e0 = __expf(a - mnew);
      float e1 = __expf(b - mnew);
      float rs = e0 + e1;
#pragma unroll
      for (int off = 1; off < 16; off <<= 1) rs += __shfl_xor(rs, off, 32);
      lrun[r] = lrun[r] * alpha + rs;
      mrun[r] = mnew;
#pragma unroll
      for (int nt = 0; nt < 8; ++nt) o[nt][r] *= alpha;
      p0[r] = e0; p1[r] = e1;
    }

    // stage P (C-layout -> A-layout) through wave-private LDS
    asm volatile("" ::: "memory");
#pragma unroll
    for (int r = 0; r < 8; ++r) {
      int prow = r + 8 * hi;
      pp[prow * 32 + ln]      = f2bf(p0[r]);
      pp[prow * 32 + 16 + ln] = f2bf(p1[r]);
    }
    asm volatile("" ::: "memory");
    v16bf pfrag = load_frag(pp + ln * 32 + hi * 16, pp + ln * 32 + hi * 16 + 8);

    // P(16x32) x V(32x128) using pre-transposed V (KV, HD, T)
#pragma unroll
    for (int nt = 0; nt < 8; ++nt) {
      const u16* vr = vbase + (size_t)(nt * 16 + ln) * T + jb + hi * 16;
      o[nt] = wmma_bf16(pfrag, load_frag(vr, vr + 8), o[nt]);
    }
  }

#pragma unroll
  for (int nt = 0; nt < 8; ++nt)
#pragma unroll
    for (int r = 0; r < 8; ++r) {
      int row = q0 + r + 8 * hi;
      int col = nt * 16 + ln;
      outb[(size_t)row * E + head * HD + col] = f2bf(o[nt][r] / lrun[r]);
    }
}

// ---------------- RMSNorm over D (bf16 or f32 output) ------------------------
__global__ __launch_bounds__(256)
void rmsnorm_kernel(const float* __restrict__ x, const float* __restrict__ w,
                    u16* __restrict__ yb, float* __restrict__ yf, int mode) {
  __shared__ float red[256];
  const int row = blockIdx.x;
  const int tid = threadIdx.x;
  const float* xr = x + (size_t)row * D;
  float s = 0.f;
#pragma unroll
  for (int e = 0; e < 4; ++e) { float v = xr[tid + 256 * e]; s += v * v; }
  red[tid] = s;
  __syncthreads();
  for (int st = 128; st > 0; st >>= 1) {
    if (tid < st) red[tid] += red[tid + st];
    __syncthreads();
  }
  float inv = rsqrtf(red[0] / (float)D + EPS);
#pragma unroll
  for (int e = 0; e < 4; ++e) {
    int i = tid + 256 * e;
    float v = xr[i] * inv * w[i];
    if (mode) yf[(size_t)row * D + i] = v;
    else      yb[(size_t)row * D + i] = f2bf(v);
  }
}

// ---------------- per-head RMSNorm + RoPE for Q ------------------------------
__global__ __launch_bounds__(128)
void qrope_kernel(const float* __restrict__ qbuf, const float* __restrict__ qn,
                  u16* __restrict__ qb16) {
  __shared__ float red[128];
  __shared__ float xn[128];
  const int t = blockIdx.x, head = blockIdx.y, d = threadIdx.x;
  float x = qbuf[(size_t)t * E + head * HD + d];
  red[d] = x * x;
  __syncthreads();
  for (int st = 64; st > 0; st >>= 1) {
    if (d < st) red[d] += red[d + st];
    __syncthreads();
  }
  float v = x * rsqrtf(red[0] / (float)HD + EPS) * qn[d];
  xn[d] = v;
  __syncthreads();
  int dd = d & 63;
  float f = (float)t * __expf(-(float)dd * 0.21586735f);  // ln(1e6)/64
  float partner = (d < 64) ? -xn[d + 64] : xn[d - 64];
  float outv = v * cosf(f) + partner * sinf(f);
  qb16[((size_t)head * T + t) * HD + d] = f2bf(outv);
}

// ---------------- per-head RMSNorm + RoPE for K (also writes keys output) ----
__global__ __launch_bounds__(128)
void krope_kernel(const float* __restrict__ kbuf, const float* __restrict__ kn,
                  float* __restrict__ keys_out, u16* __restrict__ kb16) {
  __shared__ float red[128];
  __shared__ float xn[128];
  const int t = blockIdx.x, kv = blockIdx.y, d = threadIdx.x;
  float x = kbuf[(size_t)t * (KV * HD) + kv * HD + d];
  red[d] = x * x;
  __syncthreads();
  for (int st = 64; st > 0; st >>= 1) {
    if (d < st) red[d] += red[d + st];
    __syncthreads();
  }
  float v = x * rsqrtf(red[0] / (float)HD + EPS) * kn[d];
  xn[d] = v;
  __syncthreads();
  int dd = d & 63;
  float f = (float)t * __expf(-(float)dd * 0.21586735f);
  float partner = (d < 64) ? -xn[d + 64] : xn[d - 64];
  float outv = v * cosf(f) + partner * sinf(f);
  size_t o = ((size_t)kv * T + t) * HD + d;
  keys_out[o] = outv;
  kb16[o] = f2bf(outv);
}

// ---------------- V: write vals output + transposed bf16 (KV, HD, T) ---------
__global__ __launch_bounds__(128)
void vtrans_kernel(const float* __restrict__ vbuf, float* __restrict__ vals_out,
                   u16* __restrict__ vT) {
  const int t = blockIdx.x, kv = blockIdx.y, d = threadIdx.x;
  float v = vbuf[(size_t)t * (KV * HD) + kv * HD + d];
  vals_out[((size_t)kv * T + t) * HD + d] = v;
  vT[((size_t)kv * HD + d) * T + t] = f2bf(v);
}

// ---------------- SwiGLU elementwise -----------------------------------------
__global__ __launch_bounds__(256)
void silumul_kernel(const float* __restrict__ g, const float* __restrict__ u,
                    u16* __restrict__ act, int n) {
  int base = blockIdx.x * 1024 + threadIdx.x;
#pragma unroll
  for (int e = 0; e < 4; ++e) {
    int i = base + 256 * e;
    if (i < n) {
      float gv = g[i];
      float sv = gv / (1.f + __expf(-gv));
      act[i] = f2bf(sv * u[i]);
    }
  }
}

// ---------------- f32 copy ---------------------------------------------------
__global__ __launch_bounds__(256)
void copy_kernel(const float* __restrict__ src, float* __restrict__ dst, int n) {
  int i = blockIdx.x * 256 + threadIdx.x;
  if (i < n) dst[i] = src[i];
}

// ---------------- host launcher ----------------------------------------------
extern "C" void kernel_launch(void* const* d_in, const int* in_sizes, int n_in,
                              void* d_out, int out_size, void* d_ws, size_t ws_size,
                              hipStream_t stream) {
  (void)in_sizes; (void)n_in; (void)out_size; (void)ws_size;
  const float* emb = (const float*)d_in[0];
  const float* ln1 = (const float*)d_in[1];
  const float* qw  = (const float*)d_in[2];
  const float* kw  = (const float*)d_in[3];
  const float* vw  = (const float*)d_in[4];
  const float* qn  = (const float*)d_in[5];
  const float* kn  = (const float*)d_in[6];
  const float* ow  = (const float*)d_in[7];
  const float* ln2 = (const float*)d_in[8];
  const float* gw  = (const float*)d_in[9];
  const float* uw  = (const float*)d_in[10];
  const float* dw  = (const float*)d_in[11];
  const float* nw  = (const float*)d_in[12];

  float* out      = (float*)d_out;
  float* hs_out   = out;
  float* keys_out = out + (size_t)T * D;
  float* vals_out = keys_out + (size_t)L * KV * T * HD;

  // workspace carve-up (all sizes are multiples of 4KB)
  char* ws = (char*)d_ws;
  size_t off = 0;
  auto carve = [&](size_t bytes) { char* p = ws + off; off += bytes; return p; };
  float* h       = (float*)carve((size_t)T * D * 4);          // residual stream
  u16*   xn      = (u16*)  carve((size_t)T * D * 2);          // normed acts bf16
  float* qbuf    = (float*)carve((size_t)T * E * 4);
  float* kbuf    = (float*)carve((size_t)T * KV * HD * 4);
  float* vbuf    = (float*)carve((size_t)T * KV * HD * 4);
  u16*   qb16    = (u16*)  carve((size_t)H * T * HD * 2);
  u16*   kb16    = (u16*)  carve((size_t)KV * T * HD * 2);
  u16*   vT16    = (u16*)  carve((size_t)KV * HD * T * 2);
  u16*   attnout = (u16*)  carve((size_t)T * E * 2);
  float* gatebuf = (float*)carve((size_t)T * FF * 4);
  float* upbuf   = (float*)carve((size_t)T * FF * 4);
  u16*   act16   = (u16*)  carve((size_t)T * FF * 2);

  const int nHD = T * D;
  copy_kernel<<<(nHD + 255) / 256, 256, 0, stream>>>(emb, h, nHD);

  for (int i = 0; i < L; ++i) {
    // ---- attention block ----
    rmsnorm_kernel<<<T, 256, 0, stream>>>(h, ln1 + (size_t)i * D, xn, nullptr, 0);
    gemm_bf16<<<dim3(E / 128, T / 128), 256, 0, stream>>>(
        xn, qw + (size_t)i * E * D, qbuf, T, E, D, 0);
    gemm_bf16<<<dim3((KV * HD) / 128, T / 128), 256, 0, stream>>>(
        xn, kw + (size_t)i * KV * HD * D, kbuf, T, KV * HD, D, 0);
    gemm_bf16<<<dim3((KV * HD) / 128, T / 128), 256, 0, stream>>>(
        xn, vw + (size_t)i * KV * HD * D, vbuf, T, KV * HD, D, 0);
    qrope_kernel<<<dim3(T, H), 128, 0, stream>>>(qbuf, qn + (size_t)i * HD, qb16);
    krope_kernel<<<dim3(T, KV), 128, 0, stream>>>(
        kbuf, kn + (size_t)i * HD, keys_out + (size_t)i * KV * T * HD, kb16);
    vtrans_kernel<<<dim3(T, KV), 128, 0, stream>>>(
        vbuf, vals_out + (size_t)i * KV * T * HD, vT16);
    attn_kernel<<<dim3(T / 128, H), 256, 0, stream>>>(qb16, kb16, vT16, attnout);
    gemm_bf16<<<dim3(D / 128, T / 128), 256, 0, stream>>>(
        attnout, ow + (size_t)i * D * E, h, T, D, E, 1);   // residual add

    // ---- MLP block ----
    rmsnorm_kernel<<<T, 256, 0, stream>>>(h, ln2 + (size_t)i * D, xn, nullptr, 0);
    gemm_bf16<<<dim3(FF / 128, T / 128), 256, 0, stream>>>(
        xn, gw + (size_t)i * FF * D, gatebuf, T, FF, D, 0);
    gemm_bf16<<<dim3(FF / 128, T / 128), 256, 0, stream>>>(
        xn, uw + (size_t)i * FF * D, upbuf, T, FF, D, 0);
    int nFF = T * FF;
    silumul_kernel<<<(nFF + 1023) / 1024, 256, 0, stream>>>(gatebuf, upbuf, act16, nFF);
    gemm_bf16<<<dim3(D / 128, T / 128), 256, 0, stream>>>(
        act16, dw + (size_t)i * D * FF, h, T, D, FF, 1);   // residual add
  }

  rmsnorm_kernel<<<T, 256, 0, stream>>>(h, nw, nullptr, hs_out, 1);
}
